// TestModule_6167573037201
// MI455X (gfx1250) — matrix operand order repaired and checked by reading, not compiled, not run on hardware
//
#include <hip/hip_runtime.h>

// ---------------------------------------------------------------------------
// Flash attention fwd (non-causal) for [B=2, S=2048, H=16, D=128] fp32.
// bf16 WMMA (v_wmma_f32_16x16x32_bf16); each wave owns 32 q-rows (two
// 16x16 A-subtiles) so every K/V B-fragment read from LDS feeds 2 WMMAs.
// Q is staged in per-wave LDS (bf16, scale folded) to keep the wave under
// 256 VGPRs (no s_set_vgpr_msb churn). Row-max via DPP ROW_XMASK; row-sum
// via a P x ones WMMA. LDS double-buffered K/V tiles (V transposed).
// ---------------------------------------------------------------------------

typedef __bf16 bf16_t;
typedef bf16_t v16bf __attribute__((ext_vector_type(16)));
typedef float  v8f   __attribute__((ext_vector_type(8)));

union BF16Frag {
    v16bf v;
    unsigned short h[16];
    unsigned u32[8];
    uint4 q[2];
};

// pack two floats -> packed bf16x2.  Prefer the packed builtin; otherwise use
// native __bf16 casts and let the backend pick v_cvt_pk_bf16_f32 / legalize.
__device__ __forceinline__ unsigned pack2(float lo, float hi) {
#if __has_builtin(__builtin_amdgcn_cvt_pk_bf16_f32)
    typedef __bf16 v2bf __attribute__((ext_vector_type(2)));
    v2bf r = __builtin_amdgcn_cvt_pk_bf16_f32(lo, hi);
    union { v2bf v; unsigned u; } c; c.v = r;
    return c.u;
#else
    typedef __bf16 v2bf __attribute__((ext_vector_type(2)));
    v2bf r;
    r[0] = (__bf16)lo;
    r[1] = (__bf16)hi;
    union { v2bf v; unsigned u; } c; c.v = r;
    return c.u;
#endif
}

// raw v_exp_f32 (no denorm-range fixup; softmax args <= 0, flush-to-zero ok)
__device__ __forceinline__ float fast_exp2(float x) {
#if __has_builtin(__builtin_amdgcn_exp2f)
    return __builtin_amdgcn_exp2f(x);
#else
    return __builtin_exp2f(x);
#endif
}

// DPP ROW_XMASK lane-xor within a 16-lane row (wave32: halves stay separate)
template <int MASK>
__device__ __forceinline__ float dpp_xor(float x) {
    union { float f; int i; } c; c.f = x;
    c.i = __builtin_amdgcn_update_dpp(0, c.i, 0x160 + MASK, 0xF, 0xF, true);
    return c.f;
}
__device__ __forceinline__ float row_max(float x) {
    x = fmaxf(x, dpp_xor<1>(x));
    x = fmaxf(x, dpp_xor<2>(x));
    x = fmaxf(x, dpp_xor<4>(x));
    x = fmaxf(x, dpp_xor<8>(x));
    return x;
}

#define B_SZ 2
#define S_SZ 2048
#define H_SZ 16
#define D_SZ 128
#define KT   32                        // k-rows per iteration
// (1/sqrt(128)) * log2(e) folded into Q so softmax runs in exp2 domain
#define QSCALE (0.08838834764831845f * 1.44269504088896341f)

__global__ __launch_bounds__(128, 4)   // 4 waves/SIMD -> <=256 VGPRs
void fa_fwd_bf16_wmma(const float* __restrict__ Q,
                      const float* __restrict__ K,
                      const float* __restrict__ V,
                      float* __restrict__ O) {
    __shared__ __align__(32) unsigned short sQ [4][32 * D_SZ];  // per-wave Q
    __shared__ __align__(32) unsigned short sK [2][KT * D_SZ];  // [krow][d]
    __shared__ __align__(32) unsigned short sVt[2][D_SZ * KT];  // [d][krow]
    __shared__ __align__(32) unsigned short sP [4][32 * KT];    // per-wave P

    const int tid  = threadIdx.x;
    const int w    = tid >> 5;          // wave id 0..3
    const int lane = tid & 31;
    const int half = lane >> 4;         // 0 or 1
    const int lp   = lane & 15;
    const int b    = blockIdx.z;
    const int h    = blockIdx.y;
    const int qBase = blockIdx.x * 128;

    // ---- one-time: stage this wave's 32 Q rows into LDS (bf16, scaled) ----
    {
        const int qRow = qBase + w * 32 + lane;
        const float4* q4 = reinterpret_cast<const float4*>(
            Q + (((size_t)b * S_SZ + qRow) * H_SZ + h) * D_SZ);
        uint4* dst = reinterpret_cast<uint4*>(&sQ[w][lane * D_SZ]);
        #pragma unroll
        for (int i = 0; i < 16; ++i) {
            float4 a0 = q4[2*i], a1 = q4[2*i + 1];
            uint4 pk;
            pk.x = pack2(a0.x * QSCALE, a0.y * QSCALE);
            pk.y = pack2(a0.z * QSCALE, a0.w * QSCALE);
            pk.z = pack2(a1.x * QSCALE, a1.y * QSCALE);
            pk.w = pack2(a1.z * QSCALE, a1.w * QSCALE);
            dst[i] = pk;
        }
    }

    // ---- loader role: 128 threads move 32 elements each of K and V ----
    const int lrow = tid >> 2;          // 0..31 (k-row within tile)
    const int ld   = (tid & 3) * 32;    // 0,32,64,96 (d offset)

    auto loadTiles = [&](int buf, int kb) {
        const size_t gbase = (((size_t)b * S_SZ + kb + lrow) * H_SZ + h) * (size_t)D_SZ + ld;
        const float4* k4 = reinterpret_cast<const float4*>(K + gbase);
        const float4* v4 = reinterpret_cast<const float4*>(V + gbase);
        // K: linear [krow][d], 64B contiguous packed-bf16 stores
        uint4* kdst = reinterpret_cast<uint4*>(&sK[buf][lrow * D_SZ + ld]);
        #pragma unroll
        for (int i = 0; i < 4; ++i) {
            float4 a0 = k4[2*i], a1 = k4[2*i + 1];
            uint4 pk;
            pk.x = pack2(a0.x, a0.y); pk.y = pack2(a0.z, a0.w);
            pk.z = pack2(a1.x, a1.y); pk.w = pack2(a1.z, a1.w);
            kdst[i] = pk;
        }
        // V: transposed [d][krow] (b16 + b16_d16_hi scattered stores)
        #pragma unroll
        for (int i = 0; i < 8; ++i) {
            float4 c = v4[i];
            unsigned p0 = pack2(c.x, c.y);
            unsigned p1 = pack2(c.z, c.w);
            const int d0 = ld + i * 4;
            sVt[buf][(d0 + 0) * KT + lrow] = (unsigned short)(p0);
            sVt[buf][(d0 + 1) * KT + lrow] = (unsigned short)(p0 >> 16);
            sVt[buf][(d0 + 2) * KT + lrow] = (unsigned short)(p1);
            sVt[buf][(d0 + 3) * KT + lrow] = (unsigned short)(p1 >> 16);
        }
        // prefetch the tile two iterations ahead into L2 (global_prefetch_b8)
        if (kb + 2 * KT < S_SZ) {
            __builtin_prefetch(K + gbase + (size_t)(2 * KT * H_SZ * D_SZ), 0, 1);
            __builtin_prefetch(V + gbase + (size_t)(2 * KT * H_SZ * D_SZ), 0, 1);
        }
    };

    // ---- constant ones B-fragment for row-sum WMMA ----
    BF16Frag onesB;
    #pragma unroll
    for (int j = 0; j < 8; ++j) onesB.u32[j] = 0x3F803F80u;   // bf16 1.0 pair

    // ---- per-wave state: 16 q-rows per lane (2 subtiles x 8 D-layout rows) --
    float m[2][8], l[2][8];
    v8f   o[2][8];
    #pragma unroll
    for (int u = 0; u < 2; ++u) {
        #pragma unroll
        for (int r = 0; r < 8; ++r) { m[u][r] = -1.0e30f; l[u][r] = 0.0f; }
        #pragma unroll
        for (int t = 0; t < 8; ++t) o[u][t] = (v8f)0.0f;
    }

    loadTiles(0, 0);

    const int nIter = S_SZ / KT;        // 64
    for (int it = 0; it < nIter; ++it) {
        const int buf = it & 1;
        __syncthreads();                              // buf ready; other buf free

        // ---- S = Q K^T : per subtile, two 16x16 score tiles over 32 k-rows --
        // A-frags streamed from sQ; B layout: lane col N = lp (k-row),
        // elem e -> K-dim = half*16 + e (16 contiguous bf16 in sK).
        v8f s[2][2];
        #pragma unroll
        for (int u = 0; u < 2; ++u) { s[u][0] = (v8f)0.0f; s[u][1] = (v8f)0.0f; }
        #pragma unroll
        for (int f = 0; f < 4; ++f) {
            BF16Frag qa[2], kb[2];
            #pragma unroll
            for (int u = 0; u < 2; ++u) {
                const int row = u * 16 + lp;
                qa[u].q[0] = *reinterpret_cast<const uint4*>(
                    &sQ[w][row * D_SZ + f * 32 + half * 16]);
                qa[u].q[1] = *reinterpret_cast<const uint4*>(
                    &sQ[w][row * D_SZ + f * 32 + half * 16 + 8]);
            }
            const uint4* p0 = reinterpret_cast<const uint4*>(
                &sK[buf][lp * D_SZ + f * 32 + half * 16]);
            const uint4* p1 = reinterpret_cast<const uint4*>(
                &sK[buf][(16 + lp) * D_SZ + f * 32 + half * 16]);
            kb[0].q[0] = p0[0]; kb[0].q[1] = p0[1];
            kb[1].q[0] = p1[0]; kb[1].q[1] = p1[1];
            #pragma unroll
            for (int u = 0; u < 2; ++u) {
                s[u][0] = __builtin_amdgcn_wmma_f32_16x16x32_bf16(
                              false, qa[u].v, false, kb[0].v, (short)0, s[u][0], false, false);
                s[u][1] = __builtin_amdgcn_wmma_f32_16x16x32_bf16(
                              false, qa[u].v, false, kb[1].v, (short)0, s[u][1], false, false);
            }
        }

        // issue next tile's global loads here: latency hidden by softmax+PV
        if (it + 1 < nIter) loadTiles(buf ^ 1, (it + 1) * KT);

        // ---- online softmax (exp2 domain; scale folded into Q) ----
        #pragma unroll
        for (int u = 0; u < 2; ++u) {
            v8f av;
            #pragma unroll
            for (int r = 0; r < 8; ++r) {
                const float a  = s[u][0][r];
                const float bb = s[u][1][r];
                const float c  = row_max(fmaxf(a, bb));       // DPP, VALU-only
                const float mn    = fmaxf(m[u][r], c);
                const float alpha = fast_exp2(m[u][r] - mn);
                const float p0 = fast_exp2(a  - mn);
                const float p1 = fast_exp2(bb - mn);
                l[u][r] *= alpha;                 // row-sum added via WMMA below
                m[u][r]  = mn;
                av[r]    = alpha;
                // stage P through per-wave LDS (C-layout -> A-layout transpose)
                const int row = u * 16 + r + half * 8;
                const unsigned pp = pack2(p0, p1);
                sP[w][row * KT + lp]      = (unsigned short)(pp);
                sP[w][row * KT + 16 + lp] = (unsigned short)(pp >> 16);
            }
            #pragma unroll
            for (int t = 0; t < 8; ++t) o[u][t] *= av;        // vector rescale
        }
        asm volatile("s_wait_dscnt 0" ::: "memory");

        BF16Frag pA[2];   // P as A-matrix 16x32 per subtile: row = lp
        #pragma unroll
        for (int u = 0; u < 2; ++u) {
            pA[u].q[0] = *reinterpret_cast<const uint4*>(
                &sP[w][(u * 16 + lp) * KT + half * 16]);
            pA[u].q[1] = *reinterpret_cast<const uint4*>(
                &sP[w][(u * 16 + lp) * KT + half * 16 + 8]);
        }

        // ---- row-sum of P via WMMA: D[r][*] = sum_k P[row][k] ----
        #pragma unroll
        for (int u = 0; u < 2; ++u) {
            v8f rs = __builtin_amdgcn_wmma_f32_16x16x32_bf16(
                         false, pA[u].v, false, onesB.v, (short)0, (v8f)0.0f, false, false);
            #pragma unroll
            for (int r = 0; r < 8; ++r) l[u][r] += rs[r];
        }

        // ---- O += P V : 8 output d-tiles; each V B-frag feeds 2 WMMAs ----
        #pragma unroll
        for (int t = 0; t < 8; ++t) {
            BF16Frag vb;
            const uint4* vp = reinterpret_cast<const uint4*>(
                &sVt[buf][(t * 16 + lp) * KT + half * 16]);
            vb.q[0] = vp[0]; vb.q[1] = vp[1];
            #pragma unroll
            for (int u = 0; u < 2; ++u)
                o[u][t] = __builtin_amdgcn_wmma_f32_16x16x32_bf16(
                              false, pA[u].v, false, vb.v, (short)0, o[u][t], false, false);
        }
    }

    // ---- epilogue: normalize and store (D-layout: row=r+half*8, col=lp) ----
    #pragma unroll
    for (int u = 0; u < 2; ++u) {
        #pragma unroll
        for (int r = 0; r < 8; ++r) {
            const float inv = 1.0f / l[u][r];
            const int row = qBase + w * 32 + u * 16 + r + half * 8;
            float* orow = O + (((size_t)b * S_SZ + row) * H_SZ + h) * D_SZ;
            #pragma unroll
            for (int t = 0; t < 8; ++t)
                orow[t * 16 + lp] = o[u][t][r] * inv;
        }
    }
}

extern "C" void kernel_launch(void* const* d_in, const int* in_sizes, int n_in,
                              void* d_out, int out_size, void* d_ws, size_t ws_size,
                              hipStream_t stream) {
    (void)in_sizes; (void)n_in; (void)out_size; (void)d_ws; (void)ws_size;
    const float* q = (const float*)d_in[0];
    const float* k = (const float*)d_in[1];
    const float* v = (const float*)d_in[2];
    float* out = (float*)d_out;
    dim3 grid(S_SZ / 128, H_SZ, B_SZ);   // (16, 16, 2)
    dim3 block(128);                     // 4 waves (wave32), 32 q-rows/wave
    fa_fwd_bf16_wmma<<<grid, block, 0, stream>>>(q, k, v, out);
}
